// ConditionedLSTMTransition_30305289240747
// MI455X (gfx1250) — compile-verified, matrix operand order checked by reading, not run on hardware
//
#include <hip/hip_runtime.h>
#include <hip/hip_bf16.h>
#include <stdint.h>

// Problem constants (from reference)
#define BATCH 512
#define TSTEPS 128
#define DDYNC 512
#define DSTATC 256
#define UDIMC 128
#define NOBSC 40
#define GATESC 2048 // 4*DDYN

typedef __attribute__((ext_vector_type(16))) __bf16 v16bf;
typedef __attribute__((ext_vector_type(8)))  float  v8f;
typedef __attribute__((ext_vector_type(4)))  float  v4f;

union FragU { v4f h[2]; v16bf v; };

__device__ __forceinline__ __bf16 f2bf(float f) {
  union { float f; uint32_t u; } in; in.f = f;
  uint32_t r = (in.u + 0x7FFFu + ((in.u >> 16) & 1u)) >> 16;
  union { uint16_t s; __bf16 b; } out; out.s = (uint16_t)r;
  return out.b;
}

// Load a 16x32 bf16 fragment (A row-major, or B = W^T where W rows are
// contiguous over K). Per CDNA5 16-bit A layout:
//   lanes 0-15 : row = r, K in {k0..k0+7} U {k0+16..k0+23}
//   lanes 16-31: row = r, K in {k0+8..k0+15} U {k0+24..k0+31}
__device__ __forceinline__ v16bf load_frag(const __bf16* __restrict__ base,
                                           int ld, int row, int k0, int half) {
  const __bf16* p = base + (size_t)row * ld + k0 + half * 8;
  FragU u;
  u.h[0] = *reinterpret_cast<const v4f*>(p);
  u.h[1] = *reinterpret_cast<const v4f*>(p + 16);
  return u.v;
}

__device__ __forceinline__ float sigmoidf_(float x) {
  return 1.0f / (1.0f + __expf(-x));
}

// ---------------------------------------------------------------------------
// Fused LSTM layer: GEMM (bf16 WMMA, f32 accum) + in-register cell epilogue.
//   gates[b, G*512+j] = A1[b,:] . W1[G*512+j,:]  (+ A2 . W2)  (+ Cinit)
// One wave owns a 32(batch) x 16(j) block of ALL FOUR gate groups, so each
// lane ends up with i,f,g,o for its (b,j) element in registers; the cell
// update, c in/out and h stores (f32 + bf16) happen with no gates buffer.
// K trip counts are compile-time so the scheduler fully pipelines the loop.
// Grid: (512/32)*(512/16) = 512 waves = 64 blocks of 256 threads.
// ---------------------------------------------------------------------------
template <int K1, bool HAS_A2, int K2, bool HAS_CINIT>
__global__ __launch_bounds__(256) void lstm_layer_fused_kernel(
    const __bf16* __restrict__ A1, const __bf16* __restrict__ W1,
    const __bf16* __restrict__ A2, const __bf16* __restrict__ W2,
    const float* __restrict__ Cinit,   // [B, 2048] (HAS_CINIT)
    const float* __restrict__ bias,    // [2048] (bih+bhh)
    float* __restrict__ c,             // [B, 512] in/out
    float* __restrict__ h_f32,         // [B, 512] or null (Z slice)
    __bf16* __restrict__ h_bf)         // [B, 512]
{
  const int lane = threadIdx.x & 31;
  const int wave = threadIdx.x >> 5;
  const int gw   = blockIdx.x * 8 + wave;
  const int j0   = (gw & 31) << 4;   // 32 j-blocks of 16
  const int m0   = (gw >> 5) << 5;   // 16 m-blocks of 32
  const int r    = lane & 15;
  const int half = lane >> 4;

  v8f acc[4][2];
  if (HAS_CINIT) {
#pragma unroll
    for (int G = 0; G < 4; ++G)
#pragma unroll
      for (int ti = 0; ti < 2; ++ti) {
        const int col = (G << 9) + j0 + r;
#pragma unroll
        for (int g = 0; g < 8; ++g) {
          const int row = m0 + 16 * ti + 8 * half + g;
          acc[G][ti][g] = Cinit[(size_t)row * GATESC + col];
        }
      }
  } else {
#pragma unroll
    for (int G = 0; G < 4; ++G)
#pragma unroll
      for (int ti = 0; ti < 2; ++ti)
#pragma unroll
        for (int g = 0; g < 8; ++g) acc[G][ti][g] = 0.0f;
  }

  for (int k0 = 0; k0 < K1; k0 += 32) {
    if (k0 + 32 < K1) { // warm next K-chunk (global_prefetch_b8)
      __builtin_prefetch(A1 + (size_t)(m0 + r) * K1 + k0 + 32, 0, 3);
      __builtin_prefetch(W1 + (size_t)(j0 + r) * K1 + k0 + 32, 0, 3);
    }
    v16bf a0 = load_frag(A1, K1, m0 + r,      k0, half);
    v16bf a1 = load_frag(A1, K1, m0 + 16 + r, k0, half);
#pragma unroll
    for (int G = 0; G < 4; ++G) {
      v16bf bG = load_frag(W1, K1, (G << 9) + j0 + r, k0, half);
      acc[G][0] = __builtin_amdgcn_wmma_f32_16x16x32_bf16(false, a0, false, bG, (short)0, acc[G][0], false, false);
      acc[G][1] = __builtin_amdgcn_wmma_f32_16x16x32_bf16(false, a1, false, bG, (short)0, acc[G][1], false, false);
    }
  }

  if (HAS_A2) {
    for (int k0 = 0; k0 < K2; k0 += 32) {
      v16bf a0 = load_frag(A2, K2, m0 + r,      k0, half);
      v16bf a1 = load_frag(A2, K2, m0 + 16 + r, k0, half);
#pragma unroll
      for (int G = 0; G < 4; ++G) {
        v16bf bG = load_frag(W2, K2, (G << 9) + j0 + r, k0, half);
        acc[G][0] = __builtin_amdgcn_wmma_f32_16x16x32_bf16(false, a0, false, bG, (short)0, acc[G][0], false, false);
        acc[G][1] = __builtin_amdgcn_wmma_f32_16x16x32_bf16(false, a1, false, bG, (short)0, acc[G][1], false, false);
      }
    }
  }

  // In-register LSTM cell epilogue (i,f,g,o all live in this lane).
  const int j = j0 + r;
  const float bi = bias[j];
  const float bf = bias[512 + j];
  const float bg = bias[1024 + j];
  const float bo = bias[1536 + j];
#pragma unroll
  for (int ti = 0; ti < 2; ++ti)
#pragma unroll
    for (int g = 0; g < 8; ++g) {
      const int row = m0 + 16 * ti + 8 * half + g;
      const int idx = (row << 9) + j;   // D = 512
      const float ig = acc[0][ti][g] + bi;
      const float fg = acc[1][ti][g] + bf;
      const float gg = acc[2][ti][g] + bg;
      const float og = acc[3][ti][g] + bo;
      const float cn = sigmoidf_(fg) * c[idx] + sigmoidf_(ig) * tanhf(gg);
      const float hv = sigmoidf_(og) * tanhf(cn);
      c[idx] = cn;
      if (h_f32) h_f32[idx] = hv;
      h_bf[idx] = f2bf(hv);
    }
}

// ---------------------------------------------------------------------------
// Generic bf16 WMMA GEMM (used for the time-invariant Gstat precompute and
// the small output head):  Out[M,N] = A1@W1^T (+ A2@W2^T) (+ Cinit)
// 8 waves/block, each wave a 32x32 block (2x2 WMMA tiles).
// ---------------------------------------------------------------------------
template <int K1, bool HAS_A2, int K2>
__global__ __launch_bounds__(256) void wmma_gemm_kernel(
    const __bf16* __restrict__ A1, const __bf16* __restrict__ W1,
    const __bf16* __restrict__ A2, const __bf16* __restrict__ W2,
    const float* __restrict__ Cinit, float* __restrict__ Out,
    int M, int N, int Nvalid, int ldOut)
{
  const int lane = threadIdx.x & 31;
  const int wave = threadIdx.x >> 5;
  const int gw   = blockIdx.x * 8 + wave;
  const int bnCnt = N >> 5;
  const int bm = gw / bnCnt;
  const int bn = gw - bm * bnCnt;
  const int m0 = bm << 5;
  const int n0 = bn << 5;
  if (m0 >= M) return;

  const int r    = lane & 15;
  const int half = lane >> 4;

  v8f acc[2][2];
  if (Cinit) {
#pragma unroll
    for (int ti = 0; ti < 2; ++ti)
#pragma unroll
      for (int tj = 0; tj < 2; ++tj) {
        const int col = n0 + 16 * tj + r;
#pragma unroll
        for (int g = 0; g < 8; ++g) {
          const int row = m0 + 16 * ti + 8 * half + g;
          acc[ti][tj][g] = Cinit[(size_t)row * N + col];
        }
      }
  } else {
#pragma unroll
    for (int ti = 0; ti < 2; ++ti)
#pragma unroll
      for (int tj = 0; tj < 2; ++tj)
#pragma unroll
        for (int g = 0; g < 8; ++g) acc[ti][tj][g] = 0.0f;
  }

  for (int k0 = 0; k0 < K1; k0 += 32) {
    v16bf a0 = load_frag(A1, K1, m0 + r,      k0, half);
    v16bf a1 = load_frag(A1, K1, m0 + 16 + r, k0, half);
    v16bf b0 = load_frag(W1, K1, n0 + r,      k0, half);
    v16bf b1 = load_frag(W1, K1, n0 + 16 + r, k0, half);
    acc[0][0] = __builtin_amdgcn_wmma_f32_16x16x32_bf16(false, a0, false, b0, (short)0, acc[0][0], false, false);
    acc[0][1] = __builtin_amdgcn_wmma_f32_16x16x32_bf16(false, a0, false, b1, (short)0, acc[0][1], false, false);
    acc[1][0] = __builtin_amdgcn_wmma_f32_16x16x32_bf16(false, a1, false, b0, (short)0, acc[1][0], false, false);
    acc[1][1] = __builtin_amdgcn_wmma_f32_16x16x32_bf16(false, a1, false, b1, (short)0, acc[1][1], false, false);
  }

  if (HAS_A2) {
    for (int k0 = 0; k0 < K2; k0 += 32) {
      v16bf a0 = load_frag(A2, K2, m0 + r,      k0, half);
      v16bf a1 = load_frag(A2, K2, m0 + 16 + r, k0, half);
      v16bf b0 = load_frag(W2, K2, n0 + r,      k0, half);
      v16bf b1 = load_frag(W2, K2, n0 + 16 + r, k0, half);
      acc[0][0] = __builtin_amdgcn_wmma_f32_16x16x32_bf16(false, a0, false, b0, (short)0, acc[0][0], false, false);
      acc[0][1] = __builtin_amdgcn_wmma_f32_16x16x32_bf16(false, a0, false, b1, (short)0, acc[0][1], false, false);
      acc[1][0] = __builtin_amdgcn_wmma_f32_16x16x32_bf16(false, a1, false, b0, (short)0, acc[1][0], false, false);
      acc[1][1] = __builtin_amdgcn_wmma_f32_16x16x32_bf16(false, a1, false, b1, (short)0, acc[1][1], false, false);
    }
  }

#pragma unroll
  for (int ti = 0; ti < 2; ++ti)
#pragma unroll
    for (int tj = 0; tj < 2; ++tj) {
      const int col = n0 + 16 * tj + r;
      if (col < Nvalid) {
#pragma unroll
        for (int g = 0; g < 8; ++g) {
          const int row = m0 + 16 * ti + 8 * half + g;
          Out[(size_t)row * ldOut + col] = acc[ti][tj][g];
        }
      }
    }
}

// ---- prep kernels ----
__global__ void cast_bf_kernel(const float* __restrict__ s, __bf16* __restrict__ d, int n) {
  int i = blockIdx.x * blockDim.x + threadIdx.x;
  if (i < n) d[i] = f2bf(s[i]);
}
__global__ void addcast_bf_kernel(const float* __restrict__ a, const float* __restrict__ b,
                                  __bf16* __restrict__ d, int n) {
  int i = blockIdx.x * blockDim.x + threadIdx.x;
  if (i < n) d[i] = f2bf(a[i] + b[i]);
}
__global__ void addf_kernel(const float* __restrict__ a, const float* __restrict__ b,
                            float* __restrict__ d, int n) {
  int i = blockIdx.x * blockDim.x + threadIdx.x;
  if (i < n) d[i] = a[i] + b[i];
}
__global__ void scale_cast_kernel(const float* __restrict__ s, const float* __restrict__ dtp,
                                  __bf16* __restrict__ d, int n) {
  int i = blockIdx.x * blockDim.x + threadIdx.x;
  if (i < n) d[i] = f2bf(s[i] * dtp[0]);
}
__global__ void slice_cast_kernel(const float* __restrict__ s, __bf16* __restrict__ d,
                                  int rows, int srcld, int col0, int cols) {
  int i = blockIdx.x * blockDim.x + threadIdx.x;
  int total = rows * cols;
  if (i >= total) return;
  int rr = i / cols, cc = i - rr * cols;
  d[i] = f2bf(s[(size_t)rr * srcld + col0 + cc]);
}
__global__ void pad_cast_kernel(const float* __restrict__ s, __bf16* __restrict__ d,
                                int srows, int cols, int drows) {
  int i = blockIdx.x * blockDim.x + threadIdx.x;
  int total = drows * cols;
  if (i >= total) return;
  int rr = i / cols, cc = i - rr * cols;
  float v = (rr < srows) ? s[(size_t)rr * cols + cc] : 0.0f;
  d[i] = f2bf(v);
}
__global__ void zerof_kernel(float* __restrict__ p, int n) {
  int i = blockIdx.x * blockDim.x + threadIdx.x;
  if (i < n) p[i] = 0.0f;
}

static inline int ceil_div(int a, int b) { return (a + b - 1) / b; }

extern "C" void kernel_launch(void* const* d_in, const int* in_sizes, int n_in,
                              void* d_out, int out_size, void* d_ws, size_t ws_size,
                              hipStream_t stream) {
  (void)in_sizes; (void)n_in; (void)out_size; (void)ws_size;
  const float* z_dyn    = (const float*)d_in[0];
  const float* z_static = (const float*)d_in[1];
  const float* dt       = (const float*)d_in[2];
  const float* U        = (const float*)d_in[3];
  const float* Wih0     = (const float*)d_in[4];
  const float* Whh0     = (const float*)d_in[5];
  const float* bih0     = (const float*)d_in[6];
  const float* bhh0     = (const float*)d_in[7];
  const float* Wih1     = (const float*)d_in[8];
  const float* Whh1     = (const float*)d_in[9];
  const float* bih1     = (const float*)d_in[10];
  const float* bhh1     = (const float*)d_in[11];
  const float* Cmat     = (const float*)d_in[12];
  const float* Dmat     = (const float*)d_in[13];

  float* Z = (float*)d_out;                              // [T,B,D]
  float* Y = Z + (size_t)TSTEPS * BATCH * DDYNC;         // [T,B,NOBS]

  char* ws = (char*)d_ws;
  size_t off = 0;
  auto alloc = [&](size_t bytes) -> void* {
    void* p = ws + off;
    off = (off + bytes + 255) & ~(size_t)255;
    return p;
  };

  __bf16* Udt_bf  = (__bf16*)alloc((size_t)TSTEPS * BATCH * UDIMC * 2);
  __bf16* Whh0_bf = (__bf16*)alloc((size_t)GATESC * DDYNC * 2);
  __bf16* Wu_bf   = (__bf16*)alloc((size_t)GATESC * UDIMC * 2);
  __bf16* Wst_bf  = (__bf16*)alloc((size_t)GATESC * DSTATC * 2);
  __bf16* W1_bf   = (__bf16*)alloc((size_t)GATESC * DDYNC * 2);
  __bf16* Cy_bf   = (__bf16*)alloc((size_t)64 * DDYNC * 2);
  __bf16* Dy_bf   = (__bf16*)alloc((size_t)64 * UDIMC * 2);
  __bf16* zst_bf  = (__bf16*)alloc((size_t)BATCH * DSTATC * 2);
  __bf16* h_bf    = (__bf16*)alloc((size_t)BATCH * DDYNC * 2);
  __bf16* h1_bf   = (__bf16*)alloc((size_t)BATCH * DDYNC * 2);
  float*  Gstat   = (float*)alloc((size_t)BATCH * GATESC * 4);
  float*  cA      = (float*)alloc((size_t)BATCH * DDYNC * 4);
  float*  cB      = (float*)alloc((size_t)BATCH * DDYNC * 4);
  float*  bias0   = (float*)alloc((size_t)GATESC * 4);
  float*  bias1   = (float*)alloc((size_t)GATESC * 4);

  const int TPB  = 256;
  const int nBD  = BATCH * DDYNC;      // 262144
  const int nWhh = GATESC * DDYNC;     // 1048576
  const int nU   = TSTEPS * BATCH * UDIMC;

  // ---- one-time prep (re-run every call for determinism) ----
  zerof_kernel<<<ceil_div(nBD, TPB), TPB, 0, stream>>>(cA, nBD);
  zerof_kernel<<<ceil_div(nBD, TPB), TPB, 0, stream>>>(cB, nBD);
  cast_bf_kernel<<<ceil_div(nBD, TPB), TPB, 0, stream>>>(z_dyn, h_bf, nBD);
  cast_bf_kernel<<<ceil_div(nWhh, TPB), TPB, 0, stream>>>(Whh0, Whh0_bf, nWhh);
  slice_cast_kernel<<<ceil_div(GATESC * UDIMC, TPB), TPB, 0, stream>>>(Wih0, Wu_bf, GATESC, UDIMC + DSTATC, 0, UDIMC);
  slice_cast_kernel<<<ceil_div(GATESC * DSTATC, TPB), TPB, 0, stream>>>(Wih0, Wst_bf, GATESC, UDIMC + DSTATC, UDIMC, DSTATC);
  addcast_bf_kernel<<<ceil_div(nWhh, TPB), TPB, 0, stream>>>(Wih1, Whh1, W1_bf, nWhh);
  addf_kernel<<<ceil_div(GATESC, TPB), TPB, 0, stream>>>(bih0, bhh0, bias0, GATESC);
  addf_kernel<<<ceil_div(GATESC, TPB), TPB, 0, stream>>>(bih1, bhh1, bias1, GATESC);
  pad_cast_kernel<<<ceil_div(64 * DDYNC, TPB), TPB, 0, stream>>>(Cmat, Cy_bf, NOBSC, DDYNC, 64);
  pad_cast_kernel<<<ceil_div(64 * UDIMC, TPB), TPB, 0, stream>>>(Dmat, Dy_bf, NOBSC, UDIMC, 64);
  cast_bf_kernel<<<ceil_div(BATCH * DSTATC, TPB), TPB, 0, stream>>>(z_static, zst_bf, BATCH * DSTATC);
  scale_cast_kernel<<<ceil_div(nU, TPB), TPB, 0, stream>>>(U, dt, Udt_bf, nU);

  // Gstat = z_static @ Wstat^T  (static part of layer-0 gates, time-invariant)
  const int gstatBlocks = (BATCH / 32) * (GATESC / 32) / 8;  // 128
  wmma_gemm_kernel<DSTATC, false, 0><<<gstatBlocks, 256, 0, stream>>>(
      zst_bf, Wst_bf, nullptr, nullptr, nullptr,
      Gstat, BATCH, GATESC, GATESC, GATESC);

  const int fusedBlocks = (BATCH / 32) * (DDYNC / 16) / 8;      // 64
  const int headBlocks  = ceil_div((BATCH / 32) * (64 / 32), 8); // 4

  // ---- sequential scan over T ----
  for (int t = 0; t < TSTEPS; ++t) {
    const __bf16* Ut = Udt_bf + (size_t)t * BATCH * UDIMC;

    // layer 0: gates = h@Whh0^T + ut_dt@Wu^T + Gstat; cell fused -> h1, cA
    lstm_layer_fused_kernel<DDYNC, true, UDIMC, true><<<fusedBlocks, 256, 0, stream>>>(
        h_bf, Whh0_bf, Ut, Wu_bf, Gstat,
        bias0, cA, nullptr, h1_bf);

    // layer 1: gates = h1@(Wih1+Whh1)^T; cell fused -> h2 (-> Z slice), cB
    lstm_layer_fused_kernel<DDYNC, false, 0, false><<<fusedBlocks, 256, 0, stream>>>(
        h1_bf, W1_bf, nullptr, nullptr, nullptr,
        bias1, cB, Z + (size_t)t * nBD, h_bf);

    // head: y = h2@C^T + ut_dt@D^T  (N padded 40 -> 64)
    wmma_gemm_kernel<DDYNC, true, UDIMC><<<headBlocks, 256, 0, stream>>>(
        h_bf, Cy_bf, Ut, Dy_bf, nullptr,
        Y + (size_t)t * BATCH * NOBSC, BATCH, 64, NOBSC, NOBSC);
  }
}
